// TermAwarePoolingLite_18124761989187
// MI455X (gfx1250) — compile-verified, matrix operand order checked by reading, not run on hardware
//
#include <hip/hip_runtime.h>
#include <hip/hip_bf16.h>

typedef _Float16 half16 __attribute__((ext_vector_type(16)));
typedef float    float8 __attribute__((ext_vector_type(8)));

// D = A*B + C, 16x16x32 f16 -> f32  (CDNA5 wave32 WMMA)
#define WMMA_F16(a, b, c) \
  __builtin_amdgcn_wmma_f32_16x16x32_f16(false, (a), false, (b), (short)0, (c), false, false)

// Load a 16(M) x 32(K) operand tile in the ISA A-matrix striping from a
// row-major source: lane = row (mod 16), K elements {kb+0..7, kb+16..23},
// kb = 8*(lane/16).  B operands are fed as rows of B^T with the same pattern.
static __device__ __forceinline__ half16 load_tile_h(const _Float16* __restrict__ base,
                                                     int ld, int lane) {
  const int row = lane & 15;
  const int kb  = (lane >> 4) << 3;
  const _Float16* p = base + (size_t)row * ld + kb;
  half16 t;
#pragma unroll
  for (int j = 0; j < 8; ++j) { t[j] = p[j]; t[8 + j] = p[16 + j]; }
  return t;
}

static __device__ __forceinline__ half16 load_tile_f(const float* __restrict__ base,
                                                     int ld, int lane) {
  const int row = lane & 15;
  const int kb  = (lane >> 4) << 3;
  const float* p = base + (size_t)row * ld + kb;
  half16 t;
#pragma unroll
  for (int j = 0; j < 8; ++j) { t[j] = (_Float16)p[j]; t[8 + j] = (_Float16)p[16 + j]; }
  return t;
}

// Y = X(Mx256) @ W^T(256x256); one wave computes one 16x16 output tile.
// mode 0: write f16  out[h][row(stride=rstride)][64]   (Q / K head layout)
// mode 1: write f16  out[h][64][rstride] (transposed)  (V^T head layout)
// mode 2: write f32  out[row][256]                     (final projection)
__global__ __launch_bounds__(32)
void gemm256_wmma(const float* __restrict__ X, const float* __restrict__ W, int M,
                  int mode, int rstride,
                  _Float16* __restrict__ outH, float* __restrict__ outF)
{
  const int lane  = threadIdx.x;
  const int ntile = blockIdx.x & 15;
  const int mtile = blockIdx.x >> 4;
  const float* xb = X + (size_t)(mtile * 16) * 256;
  const float* wb = W + (size_t)(ntile * 16) * 256;   // B^T rows == W rows
  float8 acc = {};
#pragma unroll
  for (int kc = 0; kc < 8; ++kc) {
    half16 a = load_tile_f(xb + kc * 32, 256, lane);
    half16 b = load_tile_f(wb + kc * 32, 256, lane);
    acc = WMMA_F16(a, b, acc);
  }
  // C/D layout: VGPR r -> row m0+r, column n = lane%16; lanes>=16 hold rows +8.
  const int n  = ntile * 16 + (lane & 15);
  const int m0 = mtile * 16 + ((lane >> 4) << 3);
#pragma unroll
  for (int r = 0; r < 8; ++r) {
    const int m = m0 + r;
    const float v = acc[r];
    if (mode == 2) {
      outF[(size_t)m * 256 + n] = v;
    } else {
      const int h = n >> 6, dk = n & 63;
      if (mode == 0) outH[((size_t)h * rstride + m) * 64 + dk] = (_Float16)v;
      else           outH[((size_t)h * 64 + dk) * rstride + m] = (_Float16)v;
    }
  }
}

// Flash attention: one wave per (graph b, head h, 16-query tile).
// Streams 32-key blocks: S = Q K^T (4 WMMA), online softmax (f32 stats,
// 16-lane shfl_xor row reductions), P restriped via LDS, ctx += P V (4 WMMA).
__global__ __launch_bounds__(32)
void attn_flash_wmma(const _Float16* __restrict__ Qh, const _Float16* __restrict__ Kh,
                     const _Float16* __restrict__ Vt, const int* __restrict__ lens,
                     float* __restrict__ ctx, int npad)
{
  __shared__ _Float16 lds_p[16 * 32];
  const int lane = threadIdx.x;
  const int qt = blockIdx.x & 63;          // T/16 = 64 query tiles
  const int h  = (blockIdx.x >> 6) & 3;
  const int b  = blockIdx.x >> 8;

  int start = 0;
#pragma unroll
  for (int i = 0; i < 16; ++i) start += (i < b) ? lens[i] : 0;
  const int len = lens[b];

  const _Float16* qbase = Qh + ((size_t)h * 1024 + (size_t)qt * 16) * 64;
  const half16 aq0 = load_tile_h(qbase,      64, lane);   // dk 0..31
  const half16 aq1 = load_tile_h(qbase + 32, 64, lane);   // dk 32..63

  const _Float16* kbase = Kh + ((size_t)h * npad + start) * 64;
  const _Float16* vbase = Vt + ((size_t)h * 64) * (size_t)npad + start;

  float8 acc0 = {}, acc1 = {}, acc2 = {}, acc3 = {};
  float mrow[8], lrow[8];
#pragma unroll
  for (int r = 0; r < 8; ++r) { mrow[r] = -__builtin_inff(); lrow[r] = 0.f; }

  const int nblk = (len + 31) >> 5;
  for (int kb = 0; kb < nblk; ++kb) {
    const int key0 = kb << 5;
    if (kb + 1 < nblk) {                     // global_prefetch_b8 next block
      __builtin_prefetch(kbase + (size_t)(key0 + 32) * 64, 0, 1);
      __builtin_prefetch(vbase + key0 + 32, 0, 1);
    }
    // --- scores: two 16-key column tiles, dk accumulated over 2 chunks ---
    float8 s0 = {}, s1 = {};
    {
      half16 b0 = load_tile_h(kbase + (size_t)key0 * 64,              64, lane);
      half16 b1 = load_tile_h(kbase + (size_t)key0 * 64 + 32,         64, lane);
      s0 = WMMA_F16(aq0, b0, s0);
      s0 = WMMA_F16(aq1, b1, s0);
      half16 b2 = load_tile_h(kbase + (size_t)(key0 + 16) * 64,       64, lane);
      half16 b3 = load_tile_h(kbase + (size_t)(key0 + 16) * 64 + 32,  64, lane);
      s1 = WMMA_F16(aq0, b2, s1);
      s1 = WMMA_F16(aq1, b3, s1);
    }
    const bool inv0 = (key0      + (lane & 15)) >= len;
    const bool inv1 = (key0 + 16 + (lane & 15)) >= len;
    // --- online softmax per query row (row m = r + 8*(lane/16)) ---
#pragma unroll
    for (int r = 0; r < 8; ++r) {
      const float a0 = inv0 ? -1e30f : s0[r] * 0.125f;   // 1/sqrt(64)
      const float a1 = inv1 ? -1e30f : s1[r] * 0.125f;
      float t = fmaxf(a0, a1);
      t = fmaxf(t, __shfl_xor(t, 1, 32));
      t = fmaxf(t, __shfl_xor(t, 2, 32));
      t = fmaxf(t, __shfl_xor(t, 4, 32));
      t = fmaxf(t, __shfl_xor(t, 8, 32));
      const float mnew = fmaxf(mrow[r], t);
      const float corr = __expf(mrow[r] - mnew);
      mrow[r] = mnew;
      acc0[r] *= corr; acc1[r] *= corr; acc2[r] *= corr; acc3[r] *= corr;
      const float p0 = __expf(a0 - mnew);
      const float p1 = __expf(a1 - mnew);
      float ps = p0 + p1;
      ps += __shfl_xor(ps, 1, 32);
      ps += __shfl_xor(ps, 2, 32);
      ps += __shfl_xor(ps, 4, 32);
      ps += __shfl_xor(ps, 8, 32);
      lrow[r] = lrow[r] * corr + ps;
      const int m = r + ((lane >> 4) << 3);
      lds_p[m * 32 +      (lane & 15)] = (_Float16)p0;
      lds_p[m * 32 + 16 + (lane & 15)] = (_Float16)p1;
    }
    __syncthreads();   // D-layout -> A-layout restripe of P through LDS
    const half16 ap = load_tile_h(lds_p, 32, lane);       // 16 queries x 32 keys
    half16 v0 = load_tile_h(vbase +                      key0, npad, lane);
    half16 v1 = load_tile_h(vbase + (size_t)16 * npad +  key0, npad, lane);
    half16 v2 = load_tile_h(vbase + (size_t)32 * npad +  key0, npad, lane);
    half16 v3 = load_tile_h(vbase + (size_t)48 * npad +  key0, npad, lane);
    acc0 = WMMA_F16(ap, v0, acc0);
    acc1 = WMMA_F16(ap, v1, acc1);
    acc2 = WMMA_F16(ap, v2, acc2);
    acc3 = WMMA_F16(ap, v3, acc3);
    __syncthreads();
  }

  float* cout = ctx + ((size_t)b * 1024 + (size_t)qt * 16) * 256 + h * 64;
  const int c = lane & 15;
#pragma unroll
  for (int r = 0; r < 8; ++r) {
    const int m = r + ((lane >> 4) << 3);
    const float inv = 1.0f / lrow[r];
    float* o = cout + (size_t)m * 256;
    o[c]      = acc0[r] * inv;
    o[16 + c] = acc1[r] * inv;
    o[32 + c] = acc2[r] * inv;
    o[48 + c] = acc3[r] * inv;
  }
}

extern "C" void kernel_launch(void* const* d_in, const int* in_sizes, int n_in,
                              void* d_out, int out_size, void* d_ws, size_t ws_size,
                              hipStream_t stream) {
  (void)n_in; (void)out_size; (void)ws_size;
  const float* go   = (const float*)d_in[0];
  const float* node = (const float*)d_in[1];
  const float* Wq   = (const float*)d_in[2];
  const float* Wk   = (const float*)d_in[3];
  const float* Wv   = (const float*)d_in[4];
  const float* Wp   = (const float*)d_in[5];
  const int*   lens = (const int*)d_in[6];
  float* out = (float*)d_out;

  const int T = in_sizes[0] / 256;                 // 1024
  const int N = in_sizes[1] / 256;                 // 24576
  const int B = in_sizes[6];                       // 16
  const int Npad = ((N + 63) & ~63) + 64;          // slack for tile loads

  // Workspace layout (~42.5 MB):
  _Float16* Qh = (_Float16*)d_ws;                  // [4][T][64]   f16
  _Float16* Kh = Qh + (size_t)4 * T * 64;          // [4][Npad][64] f16
  _Float16* Vt = Kh + (size_t)4 * Npad * 64;       // [4][64][Npad] f16 (V^T)
  float*  ctxb = (float*)(Vt + (size_t)4 * 64 * Npad); // [B][T][256] f32

  // Q = go @ Wq^T  -> Qh[h][t][dk]
  gemm256_wmma<<<dim3((T / 16) * 16), dim3(32), 0, stream>>>(
      go, Wq, T, 0, T, Qh, nullptr);
  // K = node @ Wk^T -> Kh[h][node][dk]
  gemm256_wmma<<<dim3((N / 16) * 16), dim3(32), 0, stream>>>(
      node, Wk, N, 0, Npad, Kh, nullptr);
  // V = node @ Wv^T -> Vt[h][dk][node] (transposed for the PV WMMA B-operand)
  gemm256_wmma<<<dim3((N / 16) * 16), dim3(32), 0, stream>>>(
      node, Wv, N, 1, Npad, Vt, nullptr);
  // attention -> ctxb[b][t][h*64+dk]
  attn_flash_wmma<<<dim3(B * 4 * (T / 16)), dim3(32), 0, stream>>>(
      Qh, Kh, Vt, lens, ctxb, Npad);
  // out = ctx @ Wproj^T
  gemm256_wmma<<<dim3((B * T / 16) * 16), dim3(32), 0, stream>>>(
      ctxb, Wp, B * T, 2, 0, nullptr, out);
}
